// HybridQLSTM_65481071401316
// MI455X (gfx1250) — matrix operand order repaired and checked by reading, not compiled, not run on hardware
//
#include <hip/hip_runtime.h>

#define SEQ   512
#define BATCH 64
#define DIN   512
#define DH    512
#define NG    4
#define KTOT  (DIN + DH)   // 1024
#define NROWS (NG * DH)    // 2048

typedef _Float16 f16;
typedef __attribute__((ext_vector_type(16))) _Float16 v16h;
typedef __attribute__((ext_vector_type(8)))  _Float16 v8h;
typedef __attribute__((ext_vector_type(8)))  float    v8f;

union V16U { v16h v; v8h h[2]; };

// A-matrix (16x32 f16, M x K) per-lane load, row-major source with leading dim lda.
// Layout: lanes 0-15 hold K=[0..16), lanes 16-31 hold K=[16..32) (matches B).
__device__ __forceinline__ v16h load_a_tile(const f16* __restrict__ A, int lda,
                                            int mb, int k0, int lane) {
  int m  = mb + (lane & 15);
  int kb = k0 + ((lane >> 4) << 4);
  const f16* p = A + (size_t)m * lda + kb;
  V16U u;
  u.h[0] = *(const v8h*)(p);
  u.h[1] = *(const v8h*)(p + 8);
  return u.v;
}

// B-matrix (32x16 f16, K x N): B[k][n] = W[n][k]; W row-major [n][k].
// Lanes 0-15: column n, K=[k0..k0+16); lanes 16-31: K=[k0+16..k0+32). Contiguous 32B load.
__device__ __forceinline__ v16h load_b_tile(const f16* __restrict__ Wrow0, int ldw,
                                            int k0, int lane) {
  int n  = lane & 15;
  int kb = k0 + ((lane >> 4) << 4);
  return *(const v16h*)(Wrow0 + (size_t)n * ldw + kb);
}

__device__ __forceinline__ v8f splat8(float v) {
  v8f r = {v, v, v, v, v, v, v, v};
  return r;
}

__device__ __forceinline__ float sigf(float x) { return 1.0f / (1.0f + __expf(-x)); }

__global__ void convert_x_kernel(const float* __restrict__ x, f16* __restrict__ xh, int n) {
  for (int i = blockIdx.x * blockDim.x + threadIdx.x; i < n; i += gridDim.x * blockDim.x)
    xh[i] = (f16)x[i];
}

// Pack 4 gate weight matrices [DH][DIN+DH] fp32 into Wall [NROWS][KTOT] f16 (row = g*DH+dh),
// and biases into bias_all[NROWS] fp32. Gate order: f, i, g, o (reference order).
__global__ void convert_w_kernel(const float* __restrict__ Wf, const float* __restrict__ Wi,
                                 const float* __restrict__ Wg, const float* __restrict__ Wo,
                                 const float* __restrict__ bf, const float* __restrict__ bi,
                                 const float* __restrict__ bg, const float* __restrict__ bo,
                                 f16* __restrict__ Wall, float* __restrict__ bias_all) {
  const float* Ws[NG] = {Wf, Wi, Wg, Wo};
  const float* bs[NG] = {bf, bi, bg, bo};
  int total = NROWS * KTOT;
  for (int i = blockIdx.x * blockDim.x + threadIdx.x; i < total; i += gridDim.x * blockDim.x) {
    int k   = i % KTOT;
    int row = i / KTOT;            // g*DH + dh
    int g = row / DH, dh = row % DH;
    Wall[(size_t)row * KTOT + k] = (f16)Ws[g][(size_t)dh * KTOT + k];
  }
  for (int j = blockIdx.x * blockDim.x + threadIdx.x; j < NROWS; j += gridDim.x * blockDim.x) {
    bias_all[j] = bs[j / DH][j % DH];
  }
}

__global__ void init_state_kernel(f16* __restrict__ h0, float* __restrict__ c, int n) {
  for (int i = blockIdx.x * blockDim.x + threadIdx.x; i < n; i += gridDim.x * blockDim.x) {
    h0[i] = (f16)0.0f;
    c[i]  = 0.0f;
  }
}

// One LSTM timestep. Grid: (DH/16 = 32, BATCH/16 = 4); block = 1 wave (32 threads).
// The wave computes ALL FOUR gate 16x16 tiles for its (batch tile, dh tile):
// per K-step, one A load feeds 4 independent WMMA chains (f, i, g, o).
// Gate combine + c/h update run directly on the accumulator registers — no LDS.
__global__ __launch_bounds__(32) void lstm_step_kernel(
    const f16* __restrict__ Wall, const float* __restrict__ bias,
    const f16* __restrict__ x_t,  const f16* __restrict__ h_in,
    f16* __restrict__ h_out,      float* __restrict__ c_state,
    float* __restrict__ out_stack_t, float* __restrict__ out_h,
    float* __restrict__ out_c, int t) {
  const int lane = threadIdx.x & 31;
  const int dh0  = blockIdx.x * 16;
  const int mb   = blockIdx.y * 16;     // batch tile base
  const int n    = lane & 15;

  // Per-gate weight row base pointers (row = g*DH + dh0, k-contiguous).
  const f16* Wf_ = Wall + (size_t)(0 * DH + dh0) * KTOT;
  const f16* Wi_ = Wall + (size_t)(1 * DH + dh0) * KTOT;
  const f16* Wg_ = Wall + (size_t)(2 * DH + dh0) * KTOT;
  const f16* Wo_ = Wall + (size_t)(3 * DH + dh0) * KTOT;

  // Bias folded into the accumulators (each lane's 8 D elements share column n).
  v8f accf = splat8(bias[0 * DH + dh0 + n]);
  v8f acci = splat8(bias[1 * DH + dh0 + n]);
  v8f accg = splat8(bias[2 * DH + dh0 + n]);
  v8f acco = splat8(bias[3 * DH + dh0 + n]);

  // K = [0, DIN): input contribution from x_t [BATCH][DIN]
  #pragma unroll 2
  for (int k0 = 0; k0 < DIN; k0 += 32) {
    v16h a  = load_a_tile(x_t, DIN, mb, k0, lane);
    v16h b0 = load_b_tile(Wf_, KTOT, k0, lane);
    v16h b1 = load_b_tile(Wi_, KTOT, k0, lane);
    v16h b2 = load_b_tile(Wg_, KTOT, k0, lane);
    v16h b3 = load_b_tile(Wo_, KTOT, k0, lane);
    accf = __builtin_amdgcn_wmma_f32_16x16x32_f16(false, a, false, b0, (short)0, accf, false, false);
    acci = __builtin_amdgcn_wmma_f32_16x16x32_f16(false, a, false, b1, (short)0, acci, false, false);
    accg = __builtin_amdgcn_wmma_f32_16x16x32_f16(false, a, false, b2, (short)0, accg, false, false);
    acco = __builtin_amdgcn_wmma_f32_16x16x32_f16(false, a, false, b3, (short)0, acco, false, false);
  }
  // K = [DIN, KTOT): recurrent contribution from h_{t-1} [BATCH][DH]
  #pragma unroll 2
  for (int k0 = 0; k0 < DH; k0 += 32) {
    v16h a  = load_a_tile(h_in, DH, mb, k0, lane);
    v16h b0 = load_b_tile(Wf_, KTOT, DIN + k0, lane);
    v16h b1 = load_b_tile(Wi_, KTOT, DIN + k0, lane);
    v16h b2 = load_b_tile(Wg_, KTOT, DIN + k0, lane);
    v16h b3 = load_b_tile(Wo_, KTOT, DIN + k0, lane);
    accf = __builtin_amdgcn_wmma_f32_16x16x32_f16(false, a, false, b0, (short)0, accf, false, false);
    acci = __builtin_amdgcn_wmma_f32_16x16x32_f16(false, a, false, b1, (short)0, acci, false, false);
    accg = __builtin_amdgcn_wmma_f32_16x16x32_f16(false, a, false, b2, (short)0, accg, false, false);
    acco = __builtin_amdgcn_wmma_f32_16x16x32_f16(false, a, false, b3, (short)0, acco, false, false);
  }

  // Warm next step's x slice into cache (speculative, gfx1250 global_prefetch_b8).
  if (t + 1 < SEQ)
    __builtin_prefetch(x_t + (size_t)BATCH * DIN + (size_t)threadIdx.x * 256, 0, 1);

  // D layout: element r of a lane -> row ((lane>>4)<<3)+r, col lane&15.
  // All four gates of element (m, dh) live in matching accumulator slots: fuse directly.
  const int mrow = mb + ((lane >> 4) << 3);
  const int dh   = dh0 + n;
  const bool last = (t == SEQ - 1);
  #pragma unroll
  for (int r = 0; r < 8; ++r) {
    int m = mrow + r;
    float f  = sigf(accf[r]);
    float ii = sigf(acci[r]);
    float gv = tanhf(accg[r]);
    float o  = sigf(acco[r]);
    size_t idx = (size_t)m * DH + dh;
    float cn = f * c_state[idx] + ii * gv;
    float h  = o * tanhf(cn);
    c_state[idx]     = cn;
    out_stack_t[idx] = h;
    h_out[idx]       = (f16)h;
    if (last) { out_h[idx] = h; out_c[idx] = cn; }
  }
}

extern "C" void kernel_launch(void* const* d_in, const int* in_sizes, int n_in,
                              void* d_out, int out_size, void* d_ws, size_t ws_size,
                              hipStream_t stream) {
  (void)in_sizes; (void)n_in; (void)out_size; (void)ws_size;

  const float* x  = (const float*)d_in[0];
  const float* Wf = (const float*)d_in[1];
  const float* bf = (const float*)d_in[2];
  const float* Wi = (const float*)d_in[3];
  const float* bi = (const float*)d_in[4];
  const float* Wg = (const float*)d_in[5];
  const float* bg = (const float*)d_in[6];
  const float* Wo = (const float*)d_in[7];
  const float* bo = (const float*)d_in[8];

  // Workspace carve-up (256B aligned), ~38 MB total.
  char*  base = (char*)d_ws;
  size_t off  = 0;
  auto take = [&](size_t bytes) -> void* {
    void* p = base + off;
    off = (off + bytes + 255) & ~(size_t)255;
    return p;
  };
  f16*   xh    = (f16*)  take((size_t)SEQ * BATCH * DIN * sizeof(f16));   // 32 MB
  f16*   Wall  = (f16*)  take((size_t)NROWS * KTOT * sizeof(f16));        //  4 MB
  float* bias  = (float*)take((size_t)NROWS * sizeof(float));
  f16*   h0    = (f16*)  take((size_t)BATCH * DH * sizeof(f16));
  f16*   h1    = (f16*)  take((size_t)BATCH * DH * sizeof(f16));
  float* cbuf  = (float*)take((size_t)BATCH * DH * sizeof(float));

  float* out_stack = (float*)d_out;                                 // [SEQ][BATCH][DH]
  float* out_h     = out_stack + (size_t)SEQ * BATCH * DH;          // [BATCH][DH]
  float* out_c     = out_h + (size_t)BATCH * DH;                    // [BATCH][DH]

  convert_x_kernel<<<4096, 256, 0, stream>>>(x, xh, SEQ * BATCH * DIN);
  convert_w_kernel<<<2048, 256, 0, stream>>>(Wf, Wi, Wg, Wo, bf, bi, bg, bo, Wall, bias);
  init_state_kernel<<<128, 256, 0, stream>>>(h0, cbuf, BATCH * DH);

  for (int t = 0; t < SEQ; ++t) {
    const f16* hin  = (t & 1) ? h1 : h0;
    f16*       hout = (t & 1) ? h0 : h1;
    lstm_step_kernel<<<dim3(DH / 16, BATCH / 16), 32, 0, stream>>>(
        Wall, bias, xh + (size_t)t * BATCH * DIN, hin, hout, cbuf,
        out_stack + (size_t)t * BATCH * DH, out_h, out_c, t);
  }
}
